// kMaXTransformerLayer_20358144983089
// MI455X (gfx1250) — matrix operand order repaired and checked by reading, not compiled, not run on hardware
//
#include <hip/hip_runtime.h>
#include <hip/hip_bf16.h>
#include <math.h>
#include <cstdint>

#define DEV __device__ __forceinline__

typedef __attribute__((ext_vector_type(16))) __bf16 v16bf;
typedef __attribute__((ext_vector_type(8)))  float  v8f;
typedef __attribute__((ext_vector_type(4)))  unsigned int v4u;
typedef __attribute__((ext_vector_type(8)))  int v8i;
typedef __attribute__((ext_vector_type(4)))  int v4i;

union ABReg { v16bf v; uint4 q[2]; unsigned int u[8]; unsigned short h[16]; };

DEV float gelu_f(float x) {
  return 0.5f * x * (1.0f + erff(x * 0.7071067811865475f));
}

DEV unsigned short f2bf(float f) {
  unsigned int u = __float_as_uint(f);
  unsigned int r = u + 0x7FFFu + ((u >> 16) & 1u);
  if ((u & 0x7F800000u) == 0x7F800000u) r = u;   // inf/nan: truncate
  return (unsigned short)(r >> 16);
}

DEV float bf2f(unsigned short h) {
  return __uint_as_float(((unsigned int)h) << 16);
}

// ---------------------------------------------------------------------------
// Tensor Data Mover: DMA a [32 rows x 128 halves] bf16 panel (row stride =
// rowStrideHalves) from global memory into LDS at lds_addr.
// D# built per CDNA5 ISA section 8.3/8.4 (2D tile, data_size=2B, type=2).
// ---------------------------------------------------------------------------
DEV void tdm_load_panel(unsigned int lds_addr, const void* gptr, int rowStrideHalves) {
  unsigned long long ga = (unsigned long long)(uintptr_t)gptr;
  v4u g0 = { 1u,                                            // count=1 (valid user D#)
             lds_addr,                                      // lds byte address
             (unsigned int)(ga & 0xFFFFFFFFu),              // global_addr[31:0]
             (unsigned int)((ga >> 32) & 0x01FFFFFFu) | (2u << 30) }; // ga[56:32] | type=2
  v8i g1;
  g1[0] = 0x00010000;                    // workgroup_mask=0, data_size=1 (2 bytes)
  g1[1] = (int)0xFFFF0000u;              // tensor_dim0 = 0x7FFFFFFF (low 16 << 16)
  g1[2] = (int)0xFFFF7FFFu;              // td0[31:16] | td1[15:0]<<16
  g1[3] = (int)((128u << 16) | 0x7FFFu); // td1[31:16] | tile_dim0 = 128
  g1[4] = 32;                            // tile_dim1 = 32, tile_dim2 = 0
  g1[5] = rowStrideHalves;               // tensor_dim0_stride[31:0]
  g1[6] = 0;                             // stride0[47:32] | stride1[15:0]<<16
  g1[7] = 0;                             // stride1[47:16]
  v4i z4 = {0, 0, 0, 0};
#if __clang_major__ >= 23
  v8i z8 = {0, 0, 0, 0, 0, 0, 0, 0};
  __builtin_amdgcn_tensor_load_to_lds(g0, g1, z4, z4, z8, 0);
#else
  __builtin_amdgcn_tensor_load_to_lds(g0, g1, z4, z4, 0);
#endif
}

// ---------------------------------------------------------------------------
// WMMA GEMM: D[b][o][n] = sum_k A[b][o][k] * B[b][k][n], bf16 in, f32 out.
// N is a compile-time constant -> all global/LDS accesses use immediate
// offsets. B panels staged to LDS by the TDM (double buffered); each wave
// computes a 32(O) x 32(N) block with 2x2 v_wmma_f32_16x16x32_bf16 tiles.
// Block = 4 waves covering 128 N columns.
// ---------------------------------------------------------------------------
template <int N>
__global__ __launch_bounds__(128) void wmma_gemm_bf16(
    const unsigned short* __restrict__ A, long aBatchStride,
    const unsigned short* __restrict__ Bm, long bBatchStride,
    float* __restrict__ D, long dBatchStride, int O, int K) {
  constexpr int PANW = 128;
  __shared__ unsigned short ldsB[2][32][PANW];   // 2 x 8KB panels

  const int lane = threadIdx.x;
  const int wave = threadIdx.y;
  const int lm = lane & 15;
  const int lh = lane >> 4;
  const int nblk = blockIdx.x * PANW;
  const int o0 = blockIdx.y * 32;
  const int b  = blockIdx.z;

  const unsigned short* Ab = A + (size_t)b * aBatchStride;
  const unsigned short* Bb = Bm + (size_t)b * bBatchStride + nblk;
  float* Db = D + (size_t)b * dBatchStride + nblk;

  v8f acc[2][2];
#pragma unroll
  for (int i = 0; i < 2; ++i)
#pragma unroll
    for (int j = 0; j < 2; ++j)
#pragma unroll
      for (int r = 0; r < 8; ++r) acc[i][j][r] = 0.0f;

  const int nsteps = K >> 5;
  if (wave == 0)
    tdm_load_panel((unsigned int)(uintptr_t)(&ldsB[0][0][0]), Bb, N);

  for (int s = 0; s < nsteps; ++s) {
    const int k0 = s << 5;
    const int cur = s & 1;
    if (wave == 0) __builtin_amdgcn_s_wait_tensorcnt(0);
    __syncthreads();
    if (wave == 0 && (s + 1) < nsteps)
      tdm_load_panel((unsigned int)(uintptr_t)(&ldsB[cur ^ 1][0][0]),
                     Bb + (size_t)(k0 + 32) * N, N);

    // A operand: per lane two contiguous 8-half runs (ISA 16-bit A layout):
    // halves[0..7] = row[k0+8*lh .. +7], halves[8..15] = row[k0+16+8*lh .. +7]
    v16bf a[2];
#pragma unroll
    for (int i = 0; i < 2; ++i) {
      const unsigned short* ar = Ab + (size_t)(o0 + i * 16 + lm) * K + k0 + 8 * lh;
      ABReg t;
      t.q[0] = *(const uint4*)(ar);
      t.q[1] = *(const uint4*)(ar + 16);
      a[i] = t.v;
    }

    // B operand from LDS: half e of lane = ldsB[k0 local row 8*lh + (e<8 ? e : e+8)][col]
    v16bf bb[2];
#pragma unroll
    for (int j = 0; j < 2; ++j) {
      const unsigned short* lp = &ldsB[cur][8 * lh][wave * 32 + j * 16 + lm];
      ABReg t;
#pragma unroll
      for (int e = 0; e < 16; ++e) t.h[e] = lp[(e < 8 ? e : e + 8) * PANW];
      bb[j] = t.v;
    }

#pragma unroll
    for (int i = 0; i < 2; ++i)
#pragma unroll
      for (int j = 0; j < 2; ++j)
        acc[i][j] = __builtin_amdgcn_wmma_f32_16x16x32_bf16(
            false, a[i], false, bb[j], (short)0, acc[i][j], false, false);
    __syncthreads();
  }

#pragma unroll
  for (int i = 0; i < 2; ++i)
#pragma unroll
    for (int j = 0; j < 2; ++j) {
      float* dp = Db + (size_t)(o0 + i * 16 + 8 * lh) * N + wave * 32 + j * 16 + lm;
#pragma unroll
      for (int r = 0; r < 8; ++r) dp[r * N] = acc[i][j][r];   // immediate offsets
    }
}

// ---------------------------------------------------------------------------
// Elementwise converts
// ---------------------------------------------------------------------------
__global__ void to_bf16_k(const float* __restrict__ x, unsigned short* __restrict__ y, long n) {
  long i = (long)blockIdx.x * blockDim.x + threadIdx.x;
  if (i < n) y[i] = f2bf(x[i]);
}

__global__ void gelu_to_bf16_k(const float* __restrict__ x, unsigned short* __restrict__ y, long n) {
  long i = (long)blockIdx.x * blockDim.x + threadIdx.x;
  if (i < n) y[i] = f2bf(gelu_f(x[i]));
}

// ---------------------------------------------------------------------------
// BatchNorm (train-mode, population var) stats: one block (256 thr) per channel
// x layout [B][C][N]
// ---------------------------------------------------------------------------
__global__ void bn_stats_k(const float* __restrict__ x, int C, long N, int B,
                           float* __restrict__ meanOut, float* __restrict__ invOut, float eps) {
  __shared__ float s1[256];
  __shared__ float s2[256];
  int c = blockIdx.x, t = threadIdx.x;
  float s = 0.f, ss = 0.f;
  for (int b = 0; b < B; ++b) {
    const float* p = x + ((size_t)b * C + c) * N;
    for (long i = t; i < N; i += blockDim.x) { float v = p[i]; s += v; ss += v * v; }
  }
  s1[t] = s; s2[t] = ss;
  __syncthreads();
  for (int o = 128; o > 0; o >>= 1) {
    if (t < o) { s1[t] += s1[t + o]; s2[t] += s2[t + o]; }
    __syncthreads();
  }
  if (t == 0) {
    float inv_n = 1.0f / (float)((long)B * N);
    float m = s1[0] * inv_n;
    float var = s2[0] * inv_n - m * m;
    meanOut[c] = m;
    invOut[c] = rsqrtf(fmaxf(var, 0.0f) + eps);
  }
}

// y = act( (x-mean)*inv*g + beta  [+ res] ); writes f32 and/or bf16
__global__ void bn_apply_k(const float* __restrict__ x, const float* __restrict__ mean,
                           const float* __restrict__ inv, const float* __restrict__ gb,
                           const float* __restrict__ res, float* __restrict__ outF,
                           unsigned short* __restrict__ outB, int C, long N, long total, int act) {
  long i = (long)blockIdx.x * blockDim.x + threadIdx.x;
  if (i >= total) return;
  int c = (int)((i / N) % C);
  float y = (x[i] - mean[c]) * inv[c] * gb[c] + gb[C + c];
  if (res) y += res[i];
  if (act) y = gelu_f(y);
  if (outF) outF[i] = y;
  if (outB) outB[i] = f2bf(y);
}

// ---------------------------------------------------------------------------
// Small fp32 GEMM for the L=128 query path: Y[b][o][l] = sum_k W[o][k]*X[b][k][l]
// ---------------------------------------------------------------------------
__global__ void gemm_f32_k(const float* __restrict__ W, const float* __restrict__ X,
                           float* __restrict__ Y, int O, int K, int L, int B) {
  long i = (long)blockIdx.x * blockDim.x + threadIdx.x;
  long total = (long)B * O * L;
  if (i >= total) return;
  int l = (int)(i % L);
  int o = (int)((i / L) % O);
  int b = (int)(i / ((long)L * O));
  const float* w = W + (size_t)o * K;
  const float* xp = X + (size_t)b * K * L + l;
  float acc = 0.f;
  for (int k = 0; k < K; ++k) acc = fmaf(w[k], xp[(size_t)k * L], acc);
  Y[i] = acc;
}

// depthwise 5x5x5 conv, pad 2, bf16 input / f32 weights+out
__global__ void dwconv5_k(const unsigned short* __restrict__ in, const float* __restrict__ w,
                          float* __restrict__ out, int C, int Zd, int Hd, int Wd, int B) {
  long M = (long)Zd * Hd * Wd;
  long total = (long)B * C * M;
  long i = (long)blockIdx.x * blockDim.x + threadIdx.x;
  if (i >= total) return;
  int x = (int)(i % Wd);
  int y = (int)((i / Wd) % Hd);
  int z = (int)((i / ((long)Wd * Hd)) % Zd);
  int c = (int)((i / M) % C);
  const unsigned short* p = in + (i / M) * M;
  const float* wc = w + (size_t)c * 125;
  float acc = 0.f;
  for (int dz = -2; dz <= 2; ++dz) {
    int zz = z + dz; if ((unsigned)zz >= (unsigned)Zd) continue;
    for (int dy = -2; dy <= 2; ++dy) {
      int yy = y + dy; if ((unsigned)yy >= (unsigned)Hd) continue;
      const unsigned short* row = p + ((size_t)zz * Hd + yy) * Wd;
      const float* wr = wc + ((dz + 2) * 5 + (dy + 2)) * 5;
      for (int dx = -2; dx <= 2; ++dx) {
        int xx = x + dx; if ((unsigned)xx >= (unsigned)Wd) continue;
        acc = fmaf(bf2f(row[xx]), wr[dx + 2], acc);
      }
    }
  }
  out[i] = acc;
}

// per-position L2 normalize over C channels, emit bf16
__global__ void l2norm_bf16_k(const float* __restrict__ f, unsigned short* __restrict__ o,
                              int C, long M, int B) {
  long i = (long)blockIdx.x * blockDim.x + threadIdx.x;
  if (i >= (long)B * M) return;
  long b = i / M, m = i % M;
  const float* p = f + (size_t)b * C * M + m;
  float ss = 0.f;
  for (int c = 0; c < C; ++c) { float v = p[(size_t)c * M]; ss += v * v; }
  float s = 1.0f / fmaxf(sqrtf(ss), 1e-12f);
  unsigned short* q = o + (size_t)b * C * M + m;
  for (int c = 0; c < C; ++c) q[(size_t)c * M] = f2bf(p[(size_t)c * M] * s);
}

// argmax over L (sign of pm gamma decides max vs min; mean/beta/rsqrt don't affect order)
__global__ void argmax_l_k(const float* __restrict__ ml, const float* __restrict__ pm_gb,
                           int* __restrict__ idx, int L, long M, int B) {
  long i = (long)blockIdx.x * blockDim.x + threadIdx.x;
  if (i >= (long)B * M) return;
  long b = i / M, m = i % M;
  float sgn = (pm_gb[0] >= 0.f) ? 1.f : -1.f;
  const float* p = ml + (size_t)b * L * M + m;
  float best = -3.4e38f; int bi = 0;
  for (int l = 0; l < L; ++l) {
    float v = p[(size_t)l * M] * sgn;
    if (v > best) { best = v; bi = l; }
  }
  idx[i] = bi;
}

__global__ void fill_zero_k(float* p, long n) {
  long i = (long)blockIdx.x * blockDim.x + threadIdx.x;
  if (i < n) p[i] = 0.f;
}

// km[b][c][idx[b][m]] += pval[b][c][m]
__global__ void km_scatter_k(const float* __restrict__ pval, const int* __restrict__ idx,
                             float* __restrict__ km, int C, long M, int L, int B) {
  long i = (long)blockIdx.x * blockDim.x + threadIdx.x;
  if (i >= (long)B * M) return;
  long b = i / M, m = i % M;
  int l = idx[i];
  const float* p = pval + (size_t)b * C * M + m;
  float* q = km + (size_t)b * C * L + l;
  for (int c = 0; c < C; ++c) atomicAdd(q + (size_t)c * L, p[(size_t)c * M]);
}

// y[b][l][c] = bf16(x[b][c][l])  (kern -> A-matrix for the ml WMMA gemm)
__global__ void transpose_bf16_k(const float* __restrict__ x, unsigned short* __restrict__ y,
                                 int C, int L, int B) {
  long i = (long)blockIdx.x * blockDim.x + threadIdx.x;
  if (i >= (long)B * C * L) return;
  int c = (int)(i % C);
  int l = (int)((i / C) % L);
  int b = (int)(i / ((long)C * L));
  y[((size_t)b * L + l) * C + c] = f2bf(x[((size_t)b * C + c) * L + l]);
}

// sim[b][h][l][m] = sum_d qkv[b][h*Dh+d][l] * qkv[b][KD+h*Dh+d][m]
__global__ void attn_sim_k(const float* __restrict__ qkv, float* __restrict__ sim,
                           int H, int Dh, int L, int CH, int KD, int B) {
  long i = (long)blockIdx.x * blockDim.x + threadIdx.x;
  long total = (long)B * H * L * L;
  if (i >= total) return;
  int m = (int)(i % L);
  int l = (int)((i / L) % L);
  int h = (int)((i / ((long)L * L)) % H);
  int b = (int)(i / ((long)L * L * H));
  const float* qq = qkv + ((size_t)b * CH + h * Dh) * L + l;
  const float* qk = qkv + ((size_t)b * CH + KD + h * Dh) * L + m;
  float acc = 0.f;
  for (int d = 0; d < Dh; ++d) acc = fmaf(qq[(size_t)d * L], qk[(size_t)d * L], acc);
  sim[i] = acc;
}

__global__ void softmax_rows_k(float* __restrict__ s, long rows, int L) {
  long r = (long)blockIdx.x * blockDim.x + threadIdx.x;
  if (r >= rows) return;
  float* p = s + r * (size_t)L;
  float mx = -3.4e38f;
  for (int l = 0; l < L; ++l) mx = fmaxf(mx, p[l]);
  float sum = 0.f;
  for (int l = 0; l < L; ++l) { float e = expf(p[l] - mx); p[l] = e; sum += e; }
  float is = 1.0f / sum;
  for (int l = 0; l < L; ++l) p[l] *= is;
}

// ret[b][h*Dv+d][l] = sum_m attn[b][h][l][m]*qkv[b][VOFF+h*Dv+d][m]
__global__ void attn_out_k(const float* __restrict__ attn, const float* __restrict__ qkv,
                           float* __restrict__ ret, int H, int Dv, int L, int CH, int VOFF, int B) {
  long i = (long)blockIdx.x * blockDim.x + threadIdx.x;
  long total = (long)B * H * Dv * L;
  if (i >= total) return;
  int l = (int)(i % L);
  int d = (int)((i / L) % Dv);
  int h = (int)((i / ((long)L * Dv)) % H);
  int b = (int)(i / ((long)L * Dv * H));
  const float* ap = attn + (((size_t)b * H + h) * L + l) * L;
  const float* vp = qkv + ((size_t)b * CH + VOFF + h * Dv + d) * L;
  float acc = 0.f;
  for (int m = 0; m < L; ++m) acc = fmaf(ap[m], vp[m], acc);
  ret[((size_t)b * (H * Dv) + h * Dv + d) * L + l] = acc;
}

// ---------------------------------------------------------------------------
extern "C" void kernel_launch(void* const* d_in, const int* in_sizes, int n_in,
                              void* d_out, int out_size, void* d_ws, size_t ws_size,
                              hipStream_t stream) {
  (void)in_sizes; (void)n_in; (void)out_size; (void)ws_size;

  const int B = 2, Zd = 16, Hd = 32, Wd = 32;
  const long M = (long)Zd * Hd * Wd;            // 16384
  const int L = 128, KD = 128, VD = 256, NH = 8, FFN = 2048;
  const float EPS = 1e-3f;
  constexpr int MN = 16384;                     // compile-time N for WMMA gemms

  const float* pixel   = (const float*)d_in[0];
  const float* query   = (const float*)d_in[1];
  const float* qc1_w   = (const float*)d_in[2];
  const float* qc1_gb  = (const float*)d_in[3];
  const float* pc1_w   = (const float*)d_in[4];
  const float* pc1_gb  = (const float*)d_in[5];
  const float* qkv_w   = (const float*)d_in[6];
  const float* qkv_gb  = (const float*)d_in[7];
  const float* pv_w    = (const float*)d_in[8];
  const float* pv_gb   = (const float*)d_in[9];
  const float* sim_gb  = (const float*)d_in[10];
  const float* val_gb  = (const float*)d_in[11];
  const float* qc3_w   = (const float*)d_in[12];
  const float* qc3_gb  = (const float*)d_in[13];
  const float* km_gb   = (const float*)d_in[14];
  const float* km_w    = (const float*)d_in[15];
  const float* km_gb2  = (const float*)d_in[16];
  const float* ffn1_w  = (const float*)d_in[17];
  const float* ffn1_gb = (const float*)d_in[18];
  const float* ffn2_w  = (const float*)d_in[19];
  const float* ffn2_gb = (const float*)d_in[20];
  const float* p0_w    = (const float*)d_in[21];
  const float* p0_gb   = (const float*)d_in[22];
  const float* p1_w    = (const float*)d_in[23];
  const float* p1_gb   = (const float*)d_in[24];
  const float* p2_w    = (const float*)d_in[25];
  const float* p2_gb   = (const float*)d_in[27];  // p2_b cancels inside BN
  const float* mh_w    = (const float*)d_in[28];
  const float* mh_gb   = (const float*)d_in[29];
  const float* pm_gb   = (const float*)d_in[32];

  // ---- workspace carve ----
  char* w = (char*)d_ws;
  auto carve = [&](size_t bytes) { char* p = w; w += (bytes + 255) & ~(size_t)255; return p; };
  unsigned short* wb_pc1 = (unsigned short*)carve((size_t)256 * 512 * 2);
  unsigned short* wb_pv  = (unsigned short*)carve((size_t)256 * 256 * 2);
  unsigned short* wb_p1  = (unsigned short*)carve((size_t)256 * 256 * 2);
  unsigned short* wb_p2  = (unsigned short*)carve((size_t)128 * 256 * 2);
  unsigned short* kernT  = (unsigned short*)carve((size_t)B * 128 * 128 * 2);
  unsigned short* act0   = (unsigned short*)carve((size_t)B * 512 * M * 2); // also f0/f1
  unsigned short* pixbf  = (unsigned short*)carve((size_t)B * 256 * M * 2); // also fnorm
  float* pre   = (float*)carve((size_t)B * 256 * M * 4);                    // pre-BN / f2 / ml
  float* pval  = (float*)carve((size_t)B * 256 * M * 4);
  int*   idx   = (int*)carve((size_t)B * M * 4);
  float* meanb = (float*)carve(2048 * 4);
  float* invb  = (float*)carve(2048 * 4);
  float* qsp   = (float*)carve((size_t)B * 256 * L * 4);
  float* kern  = (float*)carve((size_t)B * 128 * L * 4);
  float* km    = (float*)carve((size_t)B * 256 * L * 4);
  float* km2   = (float*)carve((size_t)B * 256 * L * 4);
  float* qbuf  = (float*)carve((size_t)B * 256 * L * 4);
  float* qkvb  = (float*)carve((size_t)B * 512 * L * 4);
  float* simb  = (float*)carve((size_t)B * 8 * L * L * 4);
  float* retb  = (float*)carve((size_t)B * 256 * L * 4);
  float* c3o   = (float*)carve((size_t)B * 256 * L * 4);
  float* hb    = (float*)carve((size_t)B * FFN * L * 4);
  float* f2o   = (float*)carve((size_t)B * 256 * L * 4);

  unsigned short* f0bf   = act0;
  unsigned short* f1bf   = act0 + (size_t)B * 256 * M;
  unsigned short* fnorm  = pixbf;
  float* f2buf = pre;                                  // [B,128,M]
  float* mlbuf = pre + (size_t)B * 128 * M;            // [B,128,M]
  float* out   = (float*)d_out;

  auto G = [](long n, int b) { return (unsigned)((n + b - 1) / b); };
  const int T = 256;

  // --- weights/activation -> bf16 ---
  to_bf16_k<<<G(256 * 512, T), T, 0, stream>>>(pc1_w, wb_pc1, 256 * 512);
  to_bf16_k<<<G(256 * 256, T), T, 0, stream>>>(pv_w, wb_pv, 256 * 256);
  to_bf16_k<<<G(256 * 256, T), T, 0, stream>>>(p1_w, wb_p1, 256 * 256);
  to_bf16_k<<<G(128 * 256, T), T, 0, stream>>>(p2_w, wb_p2, 128 * 256);
  gelu_to_bf16_k<<<G((long)B * 512 * M, T), T, 0, stream>>>(pixel, act0, (long)B * 512 * M);

  // --- pc1: 512 -> 256 ---
  wmma_gemm_bf16<MN><<<dim3(MN / 128, 256 / 32, B), dim3(32, 4), 0, stream>>>(
      wb_pc1, 0, act0, (long)512 * M, pre, (long)256 * M, 256, 512);
  bn_stats_k<<<256, 256, 0, stream>>>(pre, 256, M, B, meanb, invb, EPS);
  bn_apply_k<<<G((long)B * 256 * M, T), T, 0, stream>>>(pre, meanb, invb, pc1_gb, nullptr,
                                                        nullptr, pixbf, 256, M, (long)B * 256 * M, 1);

  // --- pv: 256 -> 256 (pval, f32) ---
  wmma_gemm_bf16<MN><<<dim3(MN / 128, 8, B), dim3(32, 4), 0, stream>>>(
      wb_pv, 0, pixbf, (long)256 * M, pre, (long)256 * M, 256, 256);
  bn_stats_k<<<256, 256, 0, stream>>>(pre, 256, M, B, meanb, invb, EPS);
  bn_apply_k<<<G((long)B * 256 * M, T), T, 0, stream>>>(pre, meanb, invb, pv_gb, nullptr,
                                                        pval, nullptr, 256, M, (long)B * 256 * M, 0);

  // --- depthwise 5^3 conv -> f0 ---
  dwconv5_k<<<G((long)B * 256 * M, T), T, 0, stream>>>(pixbf, p0_w, pre, 256, Zd, Hd, Wd, B);
  bn_stats_k<<<256, 256, 0, stream>>>(pre, 256, M, B, meanb, invb, EPS);
  bn_apply_k<<<G((long)B * 256 * M, T), T, 0, stream>>>(pre, meanb, invb, p0_gb, nullptr,
                                                        nullptr, f0bf, 256, M, (long)B * 256 * M, 1);

  // --- p1: 256 -> 256 ---
  wmma_gemm_bf16<MN><<<dim3(MN / 128, 8, B), dim3(32, 4), 0, stream>>>(
      wb_p1, 0, f0bf, (long)256 * M, pre, (long)256 * M, 256, 256);
  bn_stats_k<<<256, 256, 0, stream>>>(pre, 256, M, B, meanb, invb, EPS);
  bn_apply_k<<<G((long)B * 256 * M, T), T, 0, stream>>>(pre, meanb, invb, p1_gb, nullptr,
                                                        nullptr, f1bf, 256, M, (long)B * 256 * M, 1);

  // --- p2: 256 -> 128, BN (bias cancels), L2 normalize ---
  wmma_gemm_bf16<MN><<<dim3(MN / 128, 4, B), dim3(32, 4), 0, stream>>>(
      wb_p2, 0, f1bf, (long)256 * M, f2buf, (long)128 * M, 128, 256);
  bn_stats_k<<<128, 256, 0, stream>>>(f2buf, 128, M, B, meanb, invb, EPS);
  bn_apply_k<<<G((long)B * 128 * M, T), T, 0, stream>>>(f2buf, meanb, invb, p2_gb, nullptr,
                                                        f2buf, nullptr, 128, M, (long)B * 128 * M, 0);
  l2norm_bf16_k<<<G((long)B * M, T), T, 0, stream>>>(f2buf, fnorm, 128, M, B);

  // --- query side: qsp = gelu(bn(qc1 x)), kern = bn(mh x qsp) ---
  gemm_f32_k<<<G((long)B * 256 * L, T), T, 0, stream>>>(qc1_w, query, qsp, 256, 256, L, B);
  bn_stats_k<<<256, 256, 0, stream>>>(qsp, 256, L, B, meanb, invb, EPS);
  bn_apply_k<<<G((long)B * 256 * L, T), T, 0, stream>>>(qsp, meanb, invb, qc1_gb, nullptr,
                                                        qsp, nullptr, 256, L, (long)B * 256 * L, 1);
  gemm_f32_k<<<G((long)B * 128 * L, T), T, 0, stream>>>(mh_w, qsp, kern, 128, 256, L, B);
  bn_stats_k<<<128, 256, 0, stream>>>(kern, 128, L, B, meanb, invb, EPS);
  bn_apply_k<<<G((long)B * 128 * L, T), T, 0, stream>>>(kern, meanb, invb, mh_gb, nullptr,
                                                        kern, nullptr, 128, L, (long)B * 128 * L, 0);
  transpose_bf16_k<<<G((long)B * 128 * L, T), T, 0, stream>>>(kern, kernT, 128, L, B);

  // --- ml = kern^T x fnorm (per-batch A), argmax, km scatter ---
  wmma_gemm_bf16<MN><<<dim3(MN / 128, 4, B), dim3(32, 4), 0, stream>>>(
      kernT, (long)128 * 128, fnorm, (long)128 * M, mlbuf, (long)128 * M, 128, 128);
  argmax_l_k<<<G((long)B * M, T), T, 0, stream>>>(mlbuf, pm_gb, idx, 128, M, B);
  fill_zero_k<<<G((long)B * 256 * L, T), T, 0, stream>>>(km, (long)B * 256 * L);
  km_scatter_k<<<G((long)B * M, T), T, 0, stream>>>(pval, idx, km, 256, M, L, B);

  // --- km path: bn, conv, bn, q = query + ... ---
  bn_stats_k<<<256, 256, 0, stream>>>(km, 256, L, B, meanb, invb, EPS);
  bn_apply_k<<<G((long)B * 256 * L, T), T, 0, stream>>>(km, meanb, invb, km_gb, nullptr,
                                                        km, nullptr, 256, L, (long)B * 256 * L, 0);
  gemm_f32_k<<<G((long)B * 256 * L, T), T, 0, stream>>>(km_w, km, km2, 256, 256, L, B);
  bn_stats_k<<<256, 256, 0, stream>>>(km2, 256, L, B, meanb, invb, EPS);
  bn_apply_k<<<G((long)B * 256 * L, T), T, 0, stream>>>(km2, meanb, invb, km_gb2, query,
                                                        qbuf, nullptr, 256, L, (long)B * 256 * L, 0);

  // --- attention ---
  gemm_f32_k<<<G((long)B * 512 * L, T), T, 0, stream>>>(qkv_w, qsp, qkvb, 512, 256, L, B);
  bn_stats_k<<<512, 256, 0, stream>>>(qkvb, 512, L, B, meanb, invb, EPS);
  bn_apply_k<<<G((long)B * 512 * L, T), T, 0, stream>>>(qkvb, meanb, invb, qkv_gb, nullptr,
                                                        qkvb, nullptr, 512, L, (long)B * 512 * L, 0);
  attn_sim_k<<<G((long)B * NH * L * L, T), T, 0, stream>>>(qkvb, simb, NH, KD / NH, L, 2 * KD + VD, KD, B);
  bn_stats_k<<<8, 256, 0, stream>>>(simb, 8, (long)L * L, B, meanb, invb, EPS);
  bn_apply_k<<<G((long)B * 8 * L * L, T), T, 0, stream>>>(simb, meanb, invb, sim_gb, nullptr,
                                                          simb, nullptr, 8, (long)L * L,
                                                          (long)B * 8 * L * L, 0);
  softmax_rows_k<<<G((long)B * NH * L, T), T, 0, stream>>>(simb, (long)B * NH * L, L);
  attn_out_k<<<G((long)B * NH * (VD / NH) * L, T), T, 0, stream>>>(simb, qkvb, retb, NH, VD / NH, L,
                                                                   2 * KD + VD, 2 * KD, B);
  bn_stats_k<<<256, 256, 0, stream>>>(retb, 256, L, B, meanb, invb, EPS);
  bn_apply_k<<<G((long)B * 256 * L, T), T, 0, stream>>>(retb, meanb, invb, val_gb, nullptr,
                                                        retb, nullptr, 256, L, (long)B * 256 * L, 1);
  gemm_f32_k<<<G((long)B * 256 * L, T), T, 0, stream>>>(qc3_w, retb, c3o, 256, 256, L, B);
  bn_stats_k<<<256, 256, 0, stream>>>(c3o, 256, L, B, meanb, invb, EPS);
  bn_apply_k<<<G((long)B * 256 * L, T), T, 0, stream>>>(c3o, meanb, invb, qc3_gb, qbuf,
                                                        qbuf, nullptr, 256, L, (long)B * 256 * L, 1);

  // --- FFN ---
  gemm_f32_k<<<G((long)B * FFN * L, T), T, 0, stream>>>(ffn1_w, qbuf, hb, FFN, 256, L, B);
  bn_stats_k<<<FFN, 256, 0, stream>>>(hb, FFN, L, B, meanb, invb, EPS);
  bn_apply_k<<<G((long)B * FFN * L, T), T, 0, stream>>>(hb, meanb, invb, ffn1_gb, nullptr,
                                                        hb, nullptr, FFN, L, (long)B * FFN * L, 1);
  gemm_f32_k<<<G((long)B * 256 * L, T), T, 0, stream>>>(ffn2_w, hb, f2o, 256, FFN, L, B);
  bn_stats_k<<<256, 256, 0, stream>>>(f2o, 256, L, B, meanb, invb, EPS);
  bn_apply_k<<<G((long)B * 256 * L, T), T, 0, stream>>>(f2o, meanb, invb, ffn2_gb, qbuf,
                                                        out, nullptr, 256, L, (long)B * 256 * L, 1);
}
